// HMoENode_37203006717998
// MI455X (gfx1250) — matrix operand order repaired and checked by reading, not compiled, not run on hardware
//
#include <hip/hip_runtime.h>
#include <hip/hip_bf16.h>

#define B_   4096
#define S_   64
#define DIN_ 64
#define D_   256
#define E_   8
#define N_   4
#define C_   3
#define H_   512

typedef __attribute__((ext_vector_type(16))) __bf16 v16bf;
typedef __attribute__((ext_vector_type(8)))  float  v8f;

__device__ __forceinline__ float gelu_f(float x) {
  return 0.5f * x * (1.0f + erff(x * 0.70710678118654752f));
}
__device__ __forceinline__ float clip5(float x) {
  return fminf(5.0f, fmaxf(-5.0f, x));
}

// ---------------------------------------------------------------------------
// WMMA fragment loaders for v_wmma_f32_16x16x32_bf16 (wave32, CDNA5 layouts)
// A (16x32 bf16), row M = lane%16:
//   lanes 0-15 : VGPR0..3 -> K {0..7}, VGPR4..7 -> K {16..23}
//   lanes 16-31: VGPR0..3 -> K {8..15}, VGPR4..7 -> K {24..31}
// => per lane: two contiguous 8-element runs at k0 + (lane>=16?8:0) and +16.
__device__ __forceinline__ v16bf load_a_frag(const __bf16* A, int ld, int k0, int lane) {
  const __bf16* p = A + (size_t)(lane & 15) * ld + k0 + ((lane >= 16) ? 8 : 0);
  v16bf a;
#pragma unroll
  for (int i = 0; i < 8; ++i) { a[i] = p[i]; a[i + 8] = p[i + 16]; }
  return a;
}
// B (32x16 bf16), col N = lane%16; same K order as A: runs at k0 + (lane>=16?8:0) and +16.
// WT is the weight stored TRANSPOSED as [N][K] (ldk = K stride) so each lane
// reads two runs of 16 contiguous bytes.
__device__ __forceinline__ v16bf load_b_frag(const __bf16* WT, int ldk, int k0, int n0, int lane) {
  const __bf16* p = WT + (size_t)(n0 + (lane & 15)) * ldk + k0 + ((lane >= 16) ? 8 : 0);
  v16bf b;
#pragma unroll
  for (int i = 0; i < 8; ++i) { b[i] = p[i]; b[i + 8] = p[i + 16]; }
  return b;
}

__device__ __forceinline__ v8f wmma_bf16(v16bf a, v16bf b, v8f c) {
  return __builtin_amdgcn_wmma_f32_16x16x32_bf16(false, a, false, b,
                                                 (short)0, c, false, false);
}

// ---------------------------------------------------------------------------
// Convert fp32 weight matrices to bf16, transposed: in [batch][rows][cols]
// -> out [batch][cols][rows].
__global__ __launch_bounds__(256) void cvt_transpose_bf16(
    const float* __restrict__ in, __bf16* __restrict__ out,
    int rows, int cols, int batch) {
  size_t total = (size_t)rows * cols * batch;
  for (size_t i = (size_t)blockIdx.x * blockDim.x + threadIdx.x; i < total;
       i += (size_t)gridDim.x * blockDim.x) {
    size_t mat = i / ((size_t)rows * cols);
    size_t rc  = i % ((size_t)rows * cols);
    int r = (int)(rc / cols), c = (int)(rc % cols);
    out[(mat * cols + c) * rows + r] = (__bf16)in[i];
  }
}

// ---------------------------------------------------------------------------
// prep: h = LN(gelu(clip(x[tok, S-1, :]) @ pw[grp] + pb[grp]); lng, lnb)
// one block (256 thr) per token; grp = tok / tokensPerGroup selects weights.
__global__ __launch_bounds__(256) void prep_kernel(
    const float* __restrict__ x, const float* __restrict__ pw,
    const float* __restrict__ pb, const float* __restrict__ lng,
    const float* __restrict__ lnb, float* __restrict__ hout,
    int tokensPerGroup) {
  __shared__ float sx[DIN_];
  __shared__ float red[256];
  const int tok = blockIdx.x;
  const int t = threadIdx.x;
  const int grp = tok / tokensPerGroup;
  if (t < DIN_) sx[t] = clip5(x[((size_t)tok * S_ + (S_ - 1)) * DIN_ + t]);
  __syncthreads();
  const float* W = pw + (size_t)grp * DIN_ * D_;
  float acc = pb[(size_t)grp * D_ + t];
#pragma unroll 8
  for (int k = 0; k < DIN_; ++k) acc += sx[k] * W[k * D_ + t];
  float g = gelu_f(acc);
  red[t] = g;
  __syncthreads();
  for (int s = 128; s > 0; s >>= 1) { if (t < s) red[t] += red[t + s]; __syncthreads(); }
  float mean = red[0] * (1.0f / D_);
  __syncthreads();
  float dv = g - mean;
  red[t] = dv * dv;
  __syncthreads();
  for (int s = 128; s > 0; s >>= 1) { if (t < s) red[t] += red[t + s]; __syncthreads(); }
  float var = red[0] * (1.0f / D_);
  float val = dv * rsqrtf(var + 1e-5f) * lng[(size_t)grp * D_ + t] + lnb[(size_t)grp * D_ + t];
  hout[(size_t)tok * D_ + t] = val;
}

// ---------------------------------------------------------------------------
// gate: scores = softmax(h @ gw[grp]); top-2; dense normalized weights (T x E)
// block = 256 threads = 32 tokens x 8 experts.
__global__ __launch_bounds__(256) void gate_kernel(
    const float* __restrict__ h, const float* __restrict__ gw,
    float* __restrict__ wgt, int tokensPerGroup) {
  __shared__ float sc[32 * E_];
  const int tl = threadIdx.x >> 3, e = threadIdx.x & 7;
  const int tok = blockIdx.x * 32 + tl;
  const int grp = tok / tokensPerGroup;
  const float* G = gw + (size_t)grp * D_ * E_;
  const float* hr = h + (size_t)tok * D_;
  float s = 0.f;
  for (int k = 0; k < D_; ++k) s += hr[k] * G[k * E_ + e];
  sc[tl * E_ + e] = s;
  __syncthreads();
  if (e == 0) {
    float p[E_]; float mx = -1e30f;
#pragma unroll
    for (int i = 0; i < E_; ++i) { p[i] = sc[tl * E_ + i]; mx = fmaxf(mx, p[i]); }
    float sum = 0.f;
#pragma unroll
    for (int i = 0; i < E_; ++i) { p[i] = expf(p[i] - mx); sum += p[i]; }
    float inv = 1.f / sum;
#pragma unroll
    for (int i = 0; i < E_; ++i) p[i] *= inv;
    int i0 = 0;
#pragma unroll
    for (int i = 1; i < E_; ++i) if (p[i] > p[i0]) i0 = i;
    int i1 = (i0 == 0) ? 1 : 0;
#pragma unroll
    for (int i = 0; i < E_; ++i) if (i != i0 && p[i] > p[i1]) i1 = i;
    float gs = 1.f / (p[i0] + p[i1]);
    float* wr = wgt + (size_t)tok * E_;
#pragma unroll
    for (int i = 0; i < E_; ++i) wr[i] = 0.f;
    wr[i0] = p[i0] * gs;
    wr[i1] = p[i1] * gs;
  }
}

// ---------------------------------------------------------------------------
// Dense-weighted MoE: out[t,:] = sum_e wgt[t,e] * (gelu(h W1[e] + b1[e]) W2[e] + b2[e])
// Block = 256 threads (8 waves) per 32-token tile: 2 row-halves x 4 column-
// waves. bf16 WMMA, f32 accumulate. K-outer loops so each A fragment feeds
// 8 (pass1) / 4 (pass2) WMMAs. Weight group = tile_t0 / tokensPerGroup.
__global__ __launch_bounds__(256) void moe_wmma_kernel(
    const float* __restrict__ h, const float* __restrict__ wgt,
    const __bf16* __restrict__ w1t, const float* __restrict__ b1,
    const __bf16* __restrict__ w2t, const float* __restrict__ b2,
    float* __restrict__ out, int tokensPerGroup) {
  __shared__ __bf16 sA[32 * D_];    // activations tile, bf16 (16 KB)
  __shared__ __bf16 sH1[32 * H_];   // hidden tile, bf16 (32 KB)
  __shared__ float  sG[32 * E_];    // gate weights tile (1 KB)

  const int t0 = blockIdx.x * 32;
  const int grp = t0 / tokensPerGroup;
  const __bf16* W1 = w1t + (size_t)grp * E_ * H_ * D_;  // [e][H][D] (transposed)
  const __bf16* W2 = w2t + (size_t)grp * E_ * D_ * H_;  // [e][D][H] (transposed)
  const float*  Bb1 = b1 + (size_t)grp * E_ * H_;
  const float*  Bb2 = b2 + (size_t)grp * E_ * D_;

  const int tid = threadIdx.x;
  for (int i = tid; i < 32 * D_; i += 256) {
    int r = i >> 8, c = i & (D_ - 1);
    sA[i] = (__bf16)h[(size_t)(t0 + r) * D_ + c];
  }
  sG[tid] = wgt[(size_t)(t0 + (tid >> 3)) * E_ + (tid & 7)];
  __syncthreads();

  const int wid = tid >> 5, lane = tid & 31;
  const int half = wid >> 2;            // which 16-token row half
  const int cw = wid & 3;               // column wave within the half
  const __bf16* sAh = sA + half * 16 * D_;
  __bf16* sH1h = sH1 + half * 16 * H_;
  const int ncol = lane & 15;
  const int rbase = (lane >= 16) ? 8 : 0;
  const int rowg = half * 16 + rbase;   // tile-row of this lane's first C row

  v8f yacc[4];
#pragma unroll
  for (int j = 0; j < 4; ++j) yacc[j] = {};

  for (int e = 0; e < E_; ++e) {
    // ---- pass 1: h1 = gelu(A @ W1[e] + b1[e]) -> LDS (bf16) ----
    const __bf16* WT1 = W1 + (size_t)e * H_ * D_;
    v8f hacc[8];
#pragma unroll
    for (int i = 0; i < 8; ++i) hacc[i] = {};
#pragma unroll 2
    for (int k0 = 0; k0 < D_; k0 += 32) {
      v16bf a = load_a_frag(sAh, D_, k0, lane);
#pragma unroll
      for (int i = 0; i < 8; ++i) {
        v16bf b = load_b_frag(WT1, D_, k0, (cw + 4 * i) * 16, lane);
        hacc[i] = wmma_bf16(a, b, hacc[i]);
      }
    }
#pragma unroll
    for (int i = 0; i < 8; ++i) {
      const int col = (cw + 4 * i) * 16 + ncol;
      const float bb = Bb1[e * H_ + col];
#pragma unroll
      for (int v = 0; v < 8; ++v)
        sH1h[(rbase + v) * H_ + col] = (__bf16)gelu_f(hacc[i][v] + bb);
    }
    __syncthreads();

    // ---- pass 2: y += gate .* (h1 @ W2[e] + b2[e]) ----
    const __bf16* WT2 = W2 + (size_t)e * D_ * H_;
    v8f acc2[4];
#pragma unroll
    for (int j = 0; j < 4; ++j) acc2[j] = {};
#pragma unroll 2
    for (int k0 = 0; k0 < H_; k0 += 32) {
      v16bf a = load_a_frag(sH1h, H_, k0, lane);
#pragma unroll
      for (int j = 0; j < 4; ++j) {
        v16bf b = load_b_frag(WT2, H_, k0, (cw * 4 + j) * 16, lane);
        acc2[j] = wmma_bf16(a, b, acc2[j]);
      }
    }
#pragma unroll
    for (int j = 0; j < 4; ++j) {
      const int col = (cw * 4 + j) * 16 + ncol;
      const float bb = Bb2[e * D_ + col];
#pragma unroll
      for (int v = 0; v < 8; ++v) {
        float gate = sG[(rowg + v) * E_ + e];
        yacc[j][v] += gate * (acc2[j][v] + bb);
      }
    }
    __syncthreads();  // before next expert overwrites sH1
  }

#pragma unroll
  for (int j = 0; j < 4; ++j) {
    const int col = (cw * 4 + j) * 16 + ncol;
#pragma unroll
    for (int v = 0; v < 8; ++v)
      out[(size_t)(t0 + rowg + v) * D_ + col] = yacc[j][v];
  }
}

// ---------------------------------------------------------------------------
// cls head: clogit[n,b,c] = feat[n,b,:] @ c_cls_w[n,:,c] + c_cls_b[n,c]
__global__ __launch_bounds__(256) void cls_kernel(
    const float* __restrict__ feat, const float* __restrict__ cw,
    const float* __restrict__ cb, float* __restrict__ clogit) {
  int idx = blockIdx.x * blockDim.x + threadIdx.x;
  if (idx >= N_ * B_ * C_) return;
  int c = idx % C_;
  int nb = idx / C_;
  int n = nb / B_;
  const float* fr = feat + (size_t)nb * D_;
  const float* W = cw + (size_t)n * D_ * C_;
  float acc = cb[n * C_ + c];
  for (int k = 0; k < D_; ++k) acc += fr[k] * W[k * C_ + c];
  clogit[idx] = acc;
}

// ---------------------------------------------------------------------------
// proj: pf[b,n,:] = feat[n,b,:] + gelu(LN(feat[n,b,:]) @ p_w[n] + p_b[n])
// (writes transposed [b][n][d] layout so the routed MoE sees (B*N, D) rows)
__global__ __launch_bounds__(256) void proj_kernel(
    const float* __restrict__ feat, const float* __restrict__ lng,
    const float* __restrict__ lnb, const float* __restrict__ pw,
    const float* __restrict__ pb, float* __restrict__ pf) {
  __shared__ float sLN[D_];
  __shared__ float red[256];
  const int nb = blockIdx.x;
  const int n = nb / B_, b = nb % B_;
  const int t = threadIdx.x;
  float f = feat[(size_t)nb * D_ + t];
  red[t] = f;
  __syncthreads();
  for (int s = 128; s > 0; s >>= 1) { if (t < s) red[t] += red[t + s]; __syncthreads(); }
  float mean = red[0] * (1.0f / D_);
  __syncthreads();
  float dv = f - mean;
  red[t] = dv * dv;
  __syncthreads();
  for (int s = 128; s > 0; s >>= 1) { if (t < s) red[t] += red[t + s]; __syncthreads(); }
  float var = red[0] * (1.0f / D_);
  sLN[t] = dv * rsqrtf(var + 1e-5f) * lng[n * D_ + t] + lnb[n * D_ + t];
  __syncthreads();
  const float* W = pw + (size_t)n * D_ * D_;
  float acc = pb[n * D_ + t];
  for (int k = 0; k < D_; ++k) acc += sLN[k] * W[(size_t)k * D_ + t];
  pf[((size_t)b * N_ + n) * D_ + t] = f + gelu_f(acc);
}

// ---------------------------------------------------------------------------
// attn: logits[b,n] = clip(x_root[b]).flat @ rt_w[:,n] + rt_b[n]; softmax over n
__global__ __launch_bounds__(256) void attn_kernel(
    const float* __restrict__ xr, const float* __restrict__ rtw,
    const float* __restrict__ rtb, float* __restrict__ attn) {
  __shared__ float red[N_][256];
  const int b = blockIdx.x, t = threadIdx.x;
  float a[N_] = {0.f, 0.f, 0.f, 0.f};
  const float* xp = xr + (size_t)b * S_ * DIN_;
  for (int k = t; k < S_ * DIN_; k += 256) {
    float v = clip5(xp[k]);
    const float* w = rtw + (size_t)k * N_;
#pragma unroll
    for (int n = 0; n < N_; ++n) a[n] += v * w[n];
  }
#pragma unroll
  for (int n = 0; n < N_; ++n) red[n][t] = a[n];
  __syncthreads();
  for (int s = 128; s > 0; s >>= 1) {
    if (t < s) {
#pragma unroll
      for (int n = 0; n < N_; ++n) red[n][t] += red[n][t + s];
    }
    __syncthreads();
  }
  if (t == 0) {
    float l[N_]; float mx = -1e30f;
#pragma unroll
    for (int n = 0; n < N_; ++n) { l[n] = red[n][0] + rtb[n]; mx = fmaxf(mx, l[n]); }
    float sum = 0.f;
#pragma unroll
    for (int n = 0; n < N_; ++n) { l[n] = expf(l[n] - mx); sum += l[n]; }
    float inv = 1.f / sum;
#pragma unroll
    for (int n = 0; n < N_; ++n) attn[(size_t)b * N_ + n] = l[n] * inv;
  }
}

// ---------------------------------------------------------------------------
// pool: pooled_feat[b,d] = sum_n routed[b,n,d]*attn[b,n];
//       pooled_logit[b,c] = sum_n clogit[n,b,c]*attn[b,n]
__global__ __launch_bounds__(256) void pool_kernel(
    const float* __restrict__ routed, const float* __restrict__ clogit,
    const float* __restrict__ attn, float* __restrict__ outl,
    float* __restrict__ outf) {
  __shared__ float sa[N_];
  const int b = blockIdx.x, t = threadIdx.x;
  if (t < N_) sa[t] = attn[(size_t)b * N_ + t];
  __syncthreads();
  float acc = 0.f;
#pragma unroll
  for (int n = 0; n < N_; ++n) acc += routed[((size_t)b * N_ + n) * D_ + t] * sa[n];
  outf[(size_t)b * D_ + t] = acc;
  if (t < C_) {
    float l = 0.f;
#pragma unroll
    for (int n = 0; n < N_; ++n) l += clogit[((size_t)n * B_ + b) * C_ + t] * sa[n];
    outl[(size_t)b * C_ + t] = l;
  }
}

// ---------------------------------------------------------------------------
extern "C" void kernel_launch(void* const* d_in, const int* in_sizes, int n_in,
                              void* d_out, int out_size, void* d_ws, size_t ws_size,
                              hipStream_t stream) {
  const float* x_root   = (const float*)d_in[0];
  const float* x_child  = (const float*)d_in[1];
  const float* c_proj_w = (const float*)d_in[2];
  const float* c_proj_b = (const float*)d_in[3];
  const float* c_ln_g   = (const float*)d_in[4];
  const float* c_ln_b   = (const float*)d_in[5];
  const float* c_gate_w = (const float*)d_in[6];
  const float* c_w1     = (const float*)d_in[7];
  const float* c_b1     = (const float*)d_in[8];
  const float* c_w2     = (const float*)d_in[9];
  const float* c_b2     = (const float*)d_in[10];
  const float* c_cls_w  = (const float*)d_in[11];
  const float* c_cls_b  = (const float*)d_in[12];
  const float* p_ln_g   = (const float*)d_in[13];
  const float* p_ln_b   = (const float*)d_in[14];
  const float* p_w      = (const float*)d_in[15];
  const float* p_b      = (const float*)d_in[16];
  const float* r_proj_w = (const float*)d_in[17];
  const float* r_proj_b = (const float*)d_in[18];
  const float* r_ln_g   = (const float*)d_in[19];
  const float* r_ln_b   = (const float*)d_in[20];
  const float* l_gate_w = (const float*)d_in[21];
  const float* l_w1     = (const float*)d_in[22];
  const float* l_b1     = (const float*)d_in[23];
  const float* l_w2     = (const float*)d_in[24];
  const float* l_b2     = (const float*)d_in[25];
  const float* m_gate_w = (const float*)d_in[26];
  const float* m_w1     = (const float*)d_in[27];
  const float* m_b1     = (const float*)d_in[28];
  const float* m_w2     = (const float*)d_in[29];
  const float* m_b2     = (const float*)d_in[30];
  const float* rt_w     = (const float*)d_in[31];
  const float* rt_b     = (const float*)d_in[32];
  (void)in_sizes; (void)n_in; (void)out_size; (void)ws_size;

  float* out       = (float*)d_out;
  float* out_logit = out;                               // B x C
  float* out_feat  = out + (size_t)B_ * C_;             // B x D
  float* out_local = out_feat + (size_t)B_ * D_;        // B x D

  // -- workspace bump allocator --
  char* base = (char*)d_ws;
  size_t off = 0;
  auto alloc = [&](size_t bytes) -> void* {
    void* p = base + off;
    off += (bytes + 255) & ~(size_t)255;
    return p;
  };
  float* h_child = (float*)alloc(sizeof(float) * (size_t)N_ * B_ * D_);
  float* h_root  = (float*)alloc(sizeof(float) * (size_t)B_ * D_);
  float* feat_ch = (float*)alloc(sizeof(float) * (size_t)N_ * B_ * D_);
  float* pf      = (float*)alloc(sizeof(float) * (size_t)B_ * N_ * D_);
  float* routed  = (float*)alloc(sizeof(float) * (size_t)B_ * N_ * D_);
  float* wgt_c   = (float*)alloc(sizeof(float) * (size_t)N_ * B_ * E_);
  float* wgt_l   = (float*)alloc(sizeof(float) * (size_t)B_ * E_);
  float* wgt_m   = (float*)alloc(sizeof(float) * (size_t)B_ * N_ * E_);
  float* clogit  = (float*)alloc(sizeof(float) * (size_t)N_ * B_ * C_);
  float* attn    = (float*)alloc(sizeof(float) * (size_t)B_ * N_);
  __bf16* c_w1t = (__bf16*)alloc(sizeof(__bf16) * (size_t)N_ * E_ * D_ * H_);
  __bf16* c_w2t = (__bf16*)alloc(sizeof(__bf16) * (size_t)N_ * E_ * H_ * D_);
  __bf16* l_w1t = (__bf16*)alloc(sizeof(__bf16) * (size_t)E_ * D_ * H_);
  __bf16* l_w2t = (__bf16*)alloc(sizeof(__bf16) * (size_t)E_ * H_ * D_);
  __bf16* m_w1t = (__bf16*)alloc(sizeof(__bf16) * (size_t)E_ * D_ * H_);
  __bf16* m_w2t = (__bf16*)alloc(sizeof(__bf16) * (size_t)E_ * H_ * D_);

  // 1) weights -> bf16, transposed to [N][K] for contiguous B-fragments
  cvt_transpose_bf16<<<2048, 256, 0, stream>>>(c_w1, c_w1t, D_, H_, N_ * E_);
  cvt_transpose_bf16<<<2048, 256, 0, stream>>>(c_w2, c_w2t, H_, D_, N_ * E_);
  cvt_transpose_bf16<<<512, 256, 0, stream>>>(l_w1, l_w1t, D_, H_, E_);
  cvt_transpose_bf16<<<512, 256, 0, stream>>>(l_w2, l_w2t, H_, D_, E_);
  cvt_transpose_bf16<<<512, 256, 0, stream>>>(m_w1, m_w1t, D_, H_, E_);
  cvt_transpose_bf16<<<512, 256, 0, stream>>>(m_w2, m_w2t, H_, D_, E_);

  // 2) prep (clip last position -> proj -> gelu -> LN)
  prep_kernel<<<N_ * B_, 256, 0, stream>>>(x_child, c_proj_w, c_proj_b,
                                           c_ln_g, c_ln_b, h_child, B_);
  prep_kernel<<<B_, 256, 0, stream>>>(x_root, r_proj_w, r_proj_b,
                                      r_ln_g, r_ln_b, h_root, B_);

  // 3) gates
  gate_kernel<<<(N_ * B_) / 32, 256, 0, stream>>>(h_child, c_gate_w, wgt_c, B_);
  gate_kernel<<<B_ / 32, 256, 0, stream>>>(h_root, l_gate_w, wgt_l, B_);

  // 4) MoE (WMMA): children feat + local context (32-token tiles)
  moe_wmma_kernel<<<(N_ * B_) / 32, 256, 0, stream>>>(
      h_child, wgt_c, c_w1t, c_b1, c_w2t, c_b2, feat_ch, B_);
  moe_wmma_kernel<<<B_ / 32, 256, 0, stream>>>(
      h_root, wgt_l, l_w1t, l_b1, l_w2t, l_b2, out_local, B_);

  // 5) cls head + proj residual
  cls_kernel<<<(N_ * B_ * C_ + 255) / 256, 256, 0, stream>>>(feat_ch, c_cls_w,
                                                             c_cls_b, clogit);
  proj_kernel<<<N_ * B_, 256, 0, stream>>>(feat_ch, p_ln_g, p_ln_b, p_w, p_b, pf);

  // 6) routed MoE over (B*N, D)
  gate_kernel<<<(B_ * N_) / 32, 256, 0, stream>>>(pf, m_gate_w, wgt_m, B_ * N_);
  moe_wmma_kernel<<<(B_ * N_) / 32, 256, 0, stream>>>(
      pf, wgt_m, m_w1t, m_b1, m_w2t, m_b2, routed, B_ * N_);

  // 7) attention over x_root + final pooling
  attn_kernel<<<B_, 256, 0, stream>>>(x_root, rt_w, rt_b, attn);
  pool_kernel<<<B_, 256, 0, stream>>>(routed, clogit, attn, out_logit, out_feat);
}
